// GraphAttentionLayer_14817637171440
// MI455X (gfx1250) — compile-verified, matrix operand order checked by reading, not run on hardware
//
#include <hip/hip_runtime.h>
#include <hip/hip_bf16.h>
#include <cmath>

typedef __attribute__((ext_vector_type(16))) _Float16 v16h;
typedef __attribute__((ext_vector_type(8)))  float    v8f;

#define LRELU_ALPHA 0.2f

// ---------------------------------------------------------------------------
// Pack W (128x64 f32, row-major) into f16 WMMA B-operand register layout.
// For 16-bit operands on CDNA5 wave32: lane L holds column n = L&15;
// element e of the v16h maps to k = 8*(L>>4) + e + (e>=8 ? 8 : 0) within a
// K=32 chunk (CDNA5 ISA "16-bit A-Matrix 16x32" layout, mirrored for B).
// Index: Wp[((c*4 + t)*32 + lane)*16 + e], c = K-chunk (4), t = N-tile (4).
// ---------------------------------------------------------------------------
__global__ __launch_bounds__(256) void pack_w_kernel(const float* __restrict__ W,
                                                     _Float16* __restrict__ Wp) {
  int idx = blockIdx.x * 256 + threadIdx.x;           // 4*4*32*16 = 8192
  if (idx >= 8192) return;
  int e    = idx & 15;
  int lane = (idx >> 4) & 31;
  int t    = (idx >> 9) & 3;
  int c    = idx >> 11;
  int hi   = lane >> 4;
  int k = 32 * c + 8 * hi + e + ((e >= 8) ? 8 : 0);   // row of W (K dim)
  int n = 16 * t + (lane & 15);                        // col of W (N dim)
  Wp[idx] = (_Float16)W[k * 64 + n];
}

// ---------------------------------------------------------------------------
// CSR row pointers from sorted row[] (segment starts).
// ---------------------------------------------------------------------------
__global__ __launch_bounds__(256) void rowptr_kernel(const int* __restrict__ row,
                                                     int* __restrict__ rowptr,
                                                     int E, int N) {
  int e = blockIdx.x * blockDim.x + threadIdx.x;
  if (e >= E) return;
  int r  = row[e];
  int rp = (e == 0) ? -1 : row[e - 1];
  for (int q = rp + 1; q <= r; ++q) rowptr[q] = e;
  if (e == E - 1) {
    for (int q = r + 1; q <= N; ++q) rowptr[q] = E;
  }
}

// ---------------------------------------------------------------------------
// Wh = h @ W via v_wmma_f32_16x16x32_f16. One wave computes a 16x64 tile:
// 4 K-chunks x 4 N-tiles = 16 WMMA ops. A converted f32->f16 on the fly
// (two contiguous 32B runs per lane per chunk); B loaded pre-packed (32B/lane).
// ---------------------------------------------------------------------------
__global__ __launch_bounds__(256) void gemm_kernel(const float* __restrict__ h,
                                                   const _Float16* __restrict__ Wp,
                                                   float* __restrict__ Wh,
                                                   int n_tiles) {
  int wave = threadIdx.x >> 5;
  int lane = threadIdx.x & 31;
  int tile = blockIdx.x * 8 + wave;
  if (tile >= n_tiles) return;
  int m  = lane & 15;
  int hi = lane >> 4;

  const float* arow = h + ((size_t)tile * 16 + m) * 128;
  const v16h* B = (const v16h*)Wp;

  v8f acc[4];
  acc[0] = (v8f){}; acc[1] = (v8f){}; acc[2] = (v8f){}; acc[3] = (v8f){};

#pragma unroll
  for (int c = 0; c < 4; ++c) {
    // A: k = 32c + 8*hi + {0..7} -> elems 0..7 ; k + 16 -> elems 8..15
    const float4* p = (const float4*)(arow + 32 * c + 8 * hi);
    float4 x0 = p[0], x1 = p[1];       // k base .. +7
    float4 x2 = p[4], x3 = p[5];       // k base+16 .. +23
    v16h a;
    a[0]  = (_Float16)x0.x; a[1]  = (_Float16)x0.y;
    a[2]  = (_Float16)x0.z; a[3]  = (_Float16)x0.w;
    a[4]  = (_Float16)x1.x; a[5]  = (_Float16)x1.y;
    a[6]  = (_Float16)x1.z; a[7]  = (_Float16)x1.w;
    a[8]  = (_Float16)x2.x; a[9]  = (_Float16)x2.y;
    a[10] = (_Float16)x2.z; a[11] = (_Float16)x2.w;
    a[12] = (_Float16)x3.x; a[13] = (_Float16)x3.y;
    a[14] = (_Float16)x3.z; a[15] = (_Float16)x3.w;

#pragma unroll
    for (int t = 0; t < 4; ++t) {
      v16h b = B[(c * 4 + t) * 32 + lane];
      acc[t] = __builtin_amdgcn_wmma_f32_16x16x32_f16(
          /*neg_a=*/false, a, /*neg_b=*/false, b,
          /*c_mod=*/(short)0, acc[t], /*reuse_a=*/false, /*reuse_b=*/false);
    }
  }

  // C/D layout: lane holds col n=m, rows v + 8*hi for v = 0..7.
#pragma unroll
  for (int t = 0; t < 4; ++t) {
#pragma unroll
    for (int v = 0; v < 8; ++v) {
      Wh[((size_t)tile * 16 + v + 8 * hi) * 64 + 16 * t + m] = acc[t][v];
    }
  }
}

// ---------------------------------------------------------------------------
// Per-node attention score vectors: Wh1 = Wh.a[:64], Wh2 = Wh.a[64:].
// Wave per node: lane reads float2 of the Wh row (coalesced 256B/wave),
// butterfly-shuffle reduction.
// ---------------------------------------------------------------------------
__global__ __launch_bounds__(256) void scores_kernel(const float* __restrict__ Wh,
                                                     const float* __restrict__ a,
                                                     float* __restrict__ Wh1,
                                                     float* __restrict__ Wh2,
                                                     int N) {
  int lane = threadIdx.x & 31;
  int node = blockIdx.x * 8 + (threadIdx.x >> 5);
  if (node >= N) return;
  float2 w = *(const float2*)(Wh + (size_t)node * 64 + 2 * lane);
  float s1 = w.x * a[2 * lane]      + w.y * a[2 * lane + 1];
  float s2 = w.x * a[64 + 2 * lane] + w.y * a[64 + 2 * lane + 1];
#pragma unroll
  for (int off = 16; off > 0; off >>= 1) {
    s1 += __shfl_xor(s1, off, 32);
    s2 += __shfl_xor(s2, off, 32);
  }
  if (lane == 0) { Wh1[node] = s1; Wh2[node] = s2; }
}

// ---------------------------------------------------------------------------
// Fused: LeakyReLU edge scores -> segment softmax -> SpMM -> ELU.
// Wave per node; 3 strided passes over its edge segment (no atomics).
// Pass 3 broadcasts per-edge attention via __shfl and gathers the Wh row
// as float2 per lane (256B coalesced, L2-resident since Wh = 25.6MB << 192MB).
// ---------------------------------------------------------------------------
__global__ __launch_bounds__(256) void gat_edge_kernel(const int* __restrict__ col,
                                                       const int* __restrict__ rowptr,
                                                       const float* __restrict__ Wh,
                                                       const float* __restrict__ Wh1,
                                                       const float* __restrict__ Wh2,
                                                       float* __restrict__ esc,
                                                       float* __restrict__ out,
                                                       int N) {
  int lane = threadIdx.x & 31;
  int node = blockIdx.x * 8 + (threadIdx.x >> 5);
  if (node >= N) return;
  int start = rowptr[node], end = rowptr[node + 1];
  float wh1 = Wh1[node];

  // Pass 1: LeakyReLU scores + segment max
  float mx = -INFINITY;
  for (int j = start + lane; j < end; j += 32) {
    float x = wh1 + Wh2[col[j]];
    float e = (x > 0.f) ? x : LRELU_ALPHA * x;
    esc[j] = e;
    mx = fmaxf(mx, e);
  }
#pragma unroll
  for (int off = 16; off > 0; off >>= 1) mx = fmaxf(mx, __shfl_xor(mx, off, 32));

  // Pass 2: exp + segment sum
  float s = 0.f;
  for (int j = start + lane; j < end; j += 32) {
    float ev = __expf(esc[j] - mx);
    esc[j] = ev;
    s += ev;
  }
#pragma unroll
  for (int off = 16; off > 0; off >>= 1) s += __shfl_xor(s, off, 32);
  float inv = (s > 0.f) ? (1.f / s) : 0.f;

  // Pass 3: weighted gather-accumulate (2 features per lane)
  float2 acc = make_float2(0.f, 0.f);
  for (int base = start; base < end; base += 32) {
    int j = base + lane;
    float att = 0.f; int c = 0;
    if (j < end) { att = esc[j] * inv; c = col[j]; }
    int cnt = min(32, end - base);
    for (int k = 0; k < cnt; ++k) {
      float ak = __shfl(att, k, 32);
      int   ck = __shfl(c,   k, 32);
      float2 w = *(const float2*)(Wh + (size_t)ck * 64 + 2 * lane);
      acc.x += ak * w.x;
      acc.y += ak * w.y;
    }
  }

  float ox = (acc.x > 0.f) ? acc.x : expm1f(acc.x);
  float oy = (acc.y > 0.f) ? acc.y : expm1f(acc.y);
  *(float2*)(out + (size_t)node * 64 + 2 * lane) = make_float2(ox, oy);
}

// ---------------------------------------------------------------------------
extern "C" void kernel_launch(void* const* d_in, const int* in_sizes, int n_in,
                              void* d_out, int out_size, void* d_ws, size_t ws_size,
                              hipStream_t stream) {
  const float* h   = (const float*)d_in[0];   // [N,128]
  const int*   row = (const int*)  d_in[1];   // [E] sorted
  const int*   col = (const int*)  d_in[2];   // [E]
  const float* W   = (const float*)d_in[3];   // [128,64]
  const float* a   = (const float*)d_in[4];   // [128,1]
  float* out = (float*)d_out;                 // [N,64]

  int N = in_sizes[0] / 128;
  int E = in_sizes[1];

  // Workspace layout (ws base is 256B aligned):
  //   Wp      : 8192 f16  (16 KB)  -- packed WMMA B operand
  //   rowptr  : N+1 ints (padded)
  //   Wh      : N*64 f32
  //   Wh1,Wh2 : N f32 each
  //   esc     : E f32
  char* ws = (char*)d_ws;
  _Float16* Wp = (_Float16*)ws;
  size_t off = 16384;
  int* rowptr = (int*)(ws + off);
  off += (((size_t)(N + 1) * sizeof(int)) + 255) & ~(size_t)255;
  float* Wh = (float*)(ws + off);
  float* Wh1 = Wh + (size_t)N * 64;
  float* Wh2 = Wh1 + N;
  float* esc = Wh2 + N;

  pack_w_kernel<<<32, 256, 0, stream>>>(W, Wp);
  rowptr_kernel<<<(E + 255) / 256, 256, 0, stream>>>(row, rowptr, E, N);

  int n_tiles = N / 16;                      // N = 100000 is a multiple of 16
  gemm_kernel<<<(n_tiles + 7) / 8, 256, 0, stream>>>(h, Wp, Wh, n_tiles);
  scores_kernel<<<(N + 7) / 8, 256, 0, stream>>>(Wh, a, Wh1, Wh2, N);
  gat_edge_kernel<<<(N + 7) / 8, 256, 0, stream>>>(col, rowptr, Wh, Wh1, Wh2, esc, out, N);
}
